// MCB_43920335569044
// MI455X (gfx1250) — matrix operand order repaired and checked by reading, not compiled, not run on hardware
//
#include <hip/hip_runtime.h>

// ---------------------------------------------------------------------------
// MCB pooling as two WMMA GEMMs with on-the-fly gathered DFT twiddles.
//   N = 16000 bins, KHALF = 8001 rfft bins, padded to PADK = 8064 (mult of 32)
//   rows = BS*SEQLEN = 1024, FEAT = 2048
// ---------------------------------------------------------------------------

typedef _Float16 v16h __attribute__((ext_vector_type(16)));
typedef _Float16 v8h  __attribute__((ext_vector_type(8)));
typedef float    v8f  __attribute__((ext_vector_type(8)));

#define NBINS   16000u
#define PADK    8064u
#define RILEN   16128u            // 2*PADK
#define ROWS    1024u
#define FEAT    2048u

// d_ws layout (bytes):
//   costab f16[16000]                       @ 0        (pad to 32768)
//   xs     f16[2][1024][2048]               @ 32768    (8,388,608 B)
//   cs     f16[2 mod][1024][2 part][8064]   @ 8421376  (66,060,288 B)
//   ri     f16[1024][16128]                 @ 74481664 (33,030,144 B)
#define OFF_XS 32768u
#define OFF_CS 8421376u
#define OFF_RI 74481664u

static __device__ __forceinline__ v8f wmma_f16(v16h a, v16h b, v8f c) {
    return __builtin_amdgcn_wmma_f32_16x16x32_f16(false, a, false, b, (short)0, c,
                                                  false, false);
}

// ---------------------------------------------------------------------------
// K0: build cos table + signed features xs[mod][row][d] = x * s   (f16)
// grid = 16384 x 256 covers 2*1024*2048 = 4,194,304 elements exactly
// ---------------------------------------------------------------------------
__global__ void k_init(const float* __restrict__ xv, const float* __restrict__ xt,
                       const float* __restrict__ sv, const float* __restrict__ st,
                       _Float16* __restrict__ costab, _Float16* __restrict__ xs)
{
    unsigned e = blockIdx.x * 256u + threadIdx.x;
    unsigned m = e >> 21, rem = e & 2097151u, d = rem & 2047u;
    float x = m ? xt[rem] : xv[rem];
    float s = m ? st[d]   : sv[d];
    xs[e] = (_Float16)(x * s);
    if (e < NBINS)
        costab[e] = (_Float16)cosf((float)e * 3.9269908169872414e-4f); // 2pi/16000
}

// ---------------------------------------------------------------------------
// K1: forward "DFT of the sketch" directly from sparse form:
//   Cos[mod][r][k] = sum_d xs[r,d] * costab[(k*h_d) % N]
//   Sin[mod][r][k] = sum_d xs[r,d] * costab[(k*h_d + 12000) % N]
// GEMM M=1024, N_k=8064 (16/tile), K=2048.  B panels built cooperatively in
// LDS in WMMA-fragment order, 4 phases of 16 K-chunks (LDS = 72 KB).
// grid = 2 mod * 504 ktiles * 8 mtile-groups = 8064 blocks of 256 (8 waves)
// ---------------------------------------------------------------------------
__global__ void k_fwd(const _Float16* __restrict__ costab_g,
                      const int* __restrict__ hv, const int* __restrict__ ht,
                      const _Float16* __restrict__ xs,
                      _Float16* __restrict__ cs)
{
    extern __shared__ char smem[];
    _Float16* ctab = (_Float16*)smem;               // 32768 B
    int*      hl   = (int*)(smem + 32768);          //  8192 B
    _Float16* pcos = (_Float16*)(smem + 40960);     // 16384 B (16 chunks frag-order)
    _Float16* psin = (_Float16*)(smem + 57344);     // 16384 B

    const unsigned bx    = blockIdx.x;
    const unsigned mod   = bx / 4032u;
    const unsigned rem   = bx - mod * 4032u;
    const unsigned ktile = rem >> 3;                // [0,504)
    const unsigned mtg   = rem & 7u;
    const unsigned tid   = threadIdx.x;
    const unsigned kbase = ktile * 16u;

    {   // stage cos table (as dwords) and hash table
        const unsigned* src = (const unsigned*)costab_g;
        unsigned* dst = (unsigned*)ctab;
        for (unsigned i = tid; i < 8000u; i += 256u) dst[i] = src[i];
        const int* h = mod ? ht : hv;
        for (unsigned i = tid; i < 2048u; i += 256u) hl[i] = h[i];
    }

    const unsigned w  = tid >> 5;
    const unsigned l  = tid & 31u;
    const unsigned li = l & 15u;
    const unsigned hi = l >> 4;
    const unsigned mt = mtg * 8u + w;               // this wave's 16-row tile
    const _Float16* Arow = xs + (size_t)mod * 2097152u
                              + (size_t)(mt * 16u + li) * 2048u;

    v8f acc_c = {0.f,0.f,0.f,0.f,0.f,0.f,0.f,0.f};
    v8f acc_s = acc_c;

    for (unsigned p = 0; p < 4u; ++p) {
        __syncthreads();
        // Build B panels in fragment order.
        // B fragment order used here (same as A): lanes 0-15 -> N=lane, K elems 0..15;
        //                         lanes 16-31 -> N=lane-16, K elems 16..31.
        for (unsigned u = tid; u < 4096u; u += 256u) {   // 4096 dword slots
            unsigned cc  = u >> 8, ll = (u >> 3) & 31u, j = u & 7u;
            unsigned bhi = ll >> 4;
            unsigned bn  = kbase + (ll & 15u);           // spectral k of this lane
            #pragma unroll
            for (unsigned half = 0; half < 2u; ++half) {
                unsigned e  = j * 2u + half;             // element in v16h
                unsigned K  = e + bhi * 16u;             // K within 32-chunk
                unsigned df = (p * 16u + cc) * 32u + K;  // feature index
                unsigned idx  = (bn * (unsigned)hl[df]) % 16000u;
                unsigned sidx = idx + 12000u; if (sidx >= 16000u) sidx -= 16000u;
                unsigned off  = (cc * 32u + ll) * 16u + e;
                pcos[off] = ctab[idx];
                psin[off] = ctab[sidx];
            }
        }
        __syncthreads();
        for (unsigned cc = 0; cc < 16u; ++cc) {
            union { v16h v; v8h h2[2]; } A, Bc, Bs;
            // A 16x32 f16 fragment: lane row = li, K = hi*16..hi*16+15
            const _Float16* ap = Arow + (p * 16u + cc) * 32u + hi * 16u;
            A.h2[0] = *(const v8h*)(ap);
            A.h2[1] = *(const v8h*)(ap + 8);
            const _Float16* bp = pcos + (cc * 32u + l) * 16u;
            Bc.h2[0] = *(const v8h*)bp;  Bc.h2[1] = *(const v8h*)(bp + 8);
            const _Float16* sp = psin + (cc * 32u + l) * 16u;
            Bs.h2[0] = *(const v8h*)sp;  Bs.h2[1] = *(const v8h*)(sp + 8);
            acc_c = wmma_f16(A.v, Bc.v, acc_c);
            acc_s = wmma_f16(A.v, Bs.v, acc_s);
        }
    }

    // D layout: lanes 0-15 (M=v), lanes 16-31 (M=v+8), N = li
    const unsigned colk = kbase + li;
    const size_t   base = (size_t)mod * (1024u * RILEN);
    #pragma unroll
    for (unsigned v = 0; v < 8u; ++v) {
        unsigned ro = mt * 16u + v + hi * 8u;
        cs[base + (size_t)ro * RILEN +         colk] = (_Float16)acc_c[v];
        cs[base + (size_t)ro * RILEN + PADK +  colk] = (_Float16)acc_s[v];
    }
}

// ---------------------------------------------------------------------------
// K2: spectral cross product + irfft weights, zero-pad K to 8064.
//   Rw[k] = w_k * CosV*CosT   (w = 1/N at k=0,8000 ; 2/N otherwise)
//   Iw[k] = -(2/N) * SinV*SinT   (minus sign folded so K3 only adds)
// ---------------------------------------------------------------------------
__global__ void k_cross(const _Float16* __restrict__ cs, _Float16* __restrict__ ri)
{
    unsigned e = blockIdx.x * 256u + threadIdx.x;   // grid covers 1024*16128 exactly
    unsigned c = e % RILEN;
    unsigned part = c / PADK;
    unsigned k = c - part * PADK;
    float out = 0.f;
    if (k <= 8000u) {
        float a = (float)cs[e];                 // modality 0
        float b = (float)cs[16515072u + e];     // modality 1
        if (part == 0u) {
            float wgt = (k == 0u || k == 8000u) ? (1.f / 16000.f) : (2.f / 16000.f);
            out = wgt * a * b;
        } else {
            out = (-2.f / 16000.f) * a * b;
        }
    }
    ri[e] = (_Float16)out;
}

// ---------------------------------------------------------------------------
// K3: inverse transform GEMM:  out[r][n] = sum_k Rw*cos((k n)%N) + Iw*sin(...)
// M=1024 (supertiles of 64 rows), N=16000 (tiles of 128), K=2*8064.
// B fragments gathered from LDS cos table; per-lane twiddle offsets
// offs[i] = ((hi*16+i)*n) mod N are K-loop-invariant, so each chunk needs only
// 16 independent add+csub+ds_load_u16 plus one serial cur0 += (32n mod N).
// A (RI) chunks double-buffered through LDS (buffer picked by arithmetic off
// a single base pointer -- no LDS pointer arrays): one barrier per chunk,
// global prefetch of chunk q+1 overlaps WMMA of chunk q.
// grid = 16 * 125 = 2000 blocks of 256 (8 waves, 4 M-subtiles each)
// ---------------------------------------------------------------------------
__global__ void k_inv(const _Float16* __restrict__ costab_g,
                      const _Float16* __restrict__ ri,
                      float* __restrict__ out)
{
    extern __shared__ char smem[];
    _Float16* ctab  = (_Float16*)smem;              // 32768 B
    _Float16* abase = (_Float16*)(smem + 32768);    // 2 x 2048 halfs (8192 B)

    const unsigned bx     = blockIdx.x;
    const unsigned msuper = bx / 125u;
    const unsigned ntile  = bx - msuper * 125u;
    const unsigned tid    = threadIdx.x;
    {
        const unsigned* src = (const unsigned*)costab_g;
        unsigned* dst = (unsigned*)ctab;
        for (unsigned i = tid; i < 8000u; i += 256u) dst[i] = src[i];
    }

    const unsigned w = tid >> 5, l = tid & 31u, li = l & 15u, hi = l >> 4;
    const unsigned n   = ntile * 128u + w * 16u + li;   // this lane's output column
    const unsigned n32 = (32u * n) % 16000u;

    // K-loop-invariant per-lane twiddle offsets (B elem i <-> K = hi*16 + i)
    unsigned offs[16];
    #pragma unroll
    for (unsigned i = 0; i < 16u; ++i) offs[i] = ((hi * 16u + i) * n) % 16000u;

    v8f acc[4];
    #pragma unroll
    for (int i = 0; i < 4; ++i) acc[i] = (v8f){0.f,0.f,0.f,0.f,0.f,0.f,0.f,0.f};

    // cooperative A-chunk load map: thread t -> row t/4, 8-half segment t%4
    const size_t gbase = (size_t)(msuper * 64u + (tid >> 2)) * RILEN + (tid & 3u) * 8u;

    // RI row layout: chunk q in [0,504) lives at halfs [q*32, q*32+32)
    // (q < 252 -> Rw/cos ; q >= 252 -> Iw/sin with +12000 table offset)
    *(v8h*)(abase + tid * 8u) = *(const v8h*)(ri + gbase);   // preload chunk 0

    unsigned cur0 = 0u;                                      // (k0*n) mod N
    for (unsigned q = 0; q < 504u; ++q) {
        __syncthreads();
        if (q + 1u < 504u)      // prefetch next chunk into the other buffer
            *(v8h*)(abase + ((q + 1u) & 1u) * 2048u + tid * 8u) =
                *(const v8h*)(ri + gbase + (q + 1u) * 32u);

        const _Float16* cbuf = abase + (q & 1u) * 2048u;
        union { v16h v; _Float16 e[16]; } B;
        #pragma unroll
        for (unsigned i = 0; i < 16u; ++i) {     // independent gathers
            unsigned idx = cur0 + offs[i];
            if (idx >= 16000u) idx -= 16000u;
            B.e[i] = ctab[idx];
        }
        #pragma unroll
        for (unsigned mt = 0; mt < 4u; ++mt) {
            union { v16h v; v8h h2[2]; } A;
            const _Float16* ap = cbuf + (mt * 16u + li) * 32u + hi * 16u;
            A.h2[0] = *(const v8h*)ap;
            A.h2[1] = *(const v8h*)(ap + 8);
            acc[mt] = wmma_f16(A.v, B.v, acc[mt]);
        }

        if (q == 251u) cur0 = 12000u;            // part boundary: k resets, sin offset
        else { cur0 += n32; if (cur0 >= 16000u) cur0 -= 16000u; }
    }

    #pragma unroll
    for (unsigned mt = 0; mt < 4u; ++mt) {
        #pragma unroll
        for (unsigned v = 0; v < 8u; ++v) {
            unsigned rowg = msuper * 64u + mt * 16u + v + hi * 8u;
            out[(size_t)rowg * 16000u + n] = acc[mt][v];
        }
    }
}

// ---------------------------------------------------------------------------
// K4: signed sqrt + per-row L2 normalize, in place on d_out. 1 block per row.
// ---------------------------------------------------------------------------
__global__ void k_norm(float* __restrict__ out)
{
    __shared__ float red[256];
    const unsigned r = blockIdx.x, tid = threadIdx.x;
    float* rowp = out + (size_t)r * 16000u;
    float ss = 0.f;
    for (unsigned i = tid; i < 16000u; i += 256u) {
        float x  = rowp[i];
        float sg = (x > 0.f) ? 1.f : ((x < 0.f) ? -1.f : 0.f);
        float y  = sg * sqrtf(fabsf(x) + 1e-12f);
        rowp[i] = y;
        ss += y * y;
    }
    red[tid] = ss;
    __syncthreads();
    for (unsigned s = 128u; s > 0u; s >>= 1) {
        if (tid < s) red[tid] += red[tid + s];
        __syncthreads();
    }
    const float inv = 1.f / fmaxf(sqrtf(red[0]), 1e-12f);
    for (unsigned i = tid; i < 16000u; i += 256u) rowp[i] *= inv;
}

// ---------------------------------------------------------------------------
extern "C" void kernel_launch(void* const* d_in, const int* in_sizes, int n_in,
                              void* d_out, int out_size, void* d_ws, size_t ws_size,
                              hipStream_t stream)
{
    (void)in_sizes; (void)n_in; (void)out_size; (void)ws_size;
    const float* xv = (const float*)d_in[0];
    const float* xt = (const float*)d_in[1];
    const float* sv = (const float*)d_in[2];
    const float* st = (const float*)d_in[3];
    const int*   hv = (const int*)d_in[4];
    const int*   ht = (const int*)d_in[5];

    char* ws = (char*)d_ws;
    _Float16* costab = (_Float16*)(ws);
    _Float16* xs     = (_Float16*)(ws + OFF_XS);
    _Float16* cs     = (_Float16*)(ws + OFF_CS);
    _Float16* ri     = (_Float16*)(ws + OFF_RI);
    float* outp = (float*)d_out;

    k_init <<<16384, 256, 0, stream>>>(xv, xt, sv, st, costab, xs);
    k_fwd  <<<8064,  256, 73728, stream>>>(costab, hv, ht, xs, cs);
    k_cross<<<64512, 256, 0, stream>>>(cs, ri);
    k_inv  <<<2000,  256, 40960, stream>>>(costab, ri, outp);
    k_norm <<<1024,  256, 0, stream>>>(outp);
}
